// Encoder_fedstar_68436008894716
// MI455X (gfx1250) — compile-verified, matrix operand order checked by reading, not run on hardware
//
#include <hip/hip_runtime.h>

typedef __attribute__((ext_vector_type(16))) _Float16 v16h;
typedef __attribute__((ext_vector_type(8)))  float    v8f;

#define DF 64            // feature dim
#define BN_EPS 1e-4f

// ---------------- WMMA helpers ----------------

__device__ __forceinline__ v8f wmma16(v16h a, v16h b, v8f c) {
  // D = A(16x32 f16) * B(32x16 f16) + C(16x16 f32)
  return __builtin_amdgcn_wmma_f32_16x16x32_f16(false, a, false, b, (short)0, c, false, false);
}

// A fragment from f32 row-major [M x lda], tile rows m0..m0+15, K-slab k0..k0+31
__device__ __forceinline__ v16h load_a_f32(const float* __restrict__ A, int lda,
                                           int m0, int k0, int lane) {
  int l = lane & 15, hi = lane >> 4;
  const float* ar = A + (size_t)(m0 + l) * lda + k0 + hi * 8;
  v16h a;
#pragma unroll
  for (int j = 0; j < 8; ++j) a[j] = (_Float16)ar[j];
#pragma unroll
  for (int j = 0; j < 8; ++j) a[8 + j] = (_Float16)ar[16 + j];
  return a;
}

// B fragment from fragment-major packed LDS: [(kslab*4+ntile)*32 + lane] * 16 halves
__device__ __forceinline__ v16h load_b_lds(const _Float16* sW, int kslab, int nt, int lane) {
  return *(const v16h*)(sW + (size_t)(((kslab * 4 + nt) * 32 + lane)) * 16);
}

// cooperative global->LDS copy in 16B chunks
__device__ __forceinline__ void stage_lds(_Float16* dst, const _Float16* __restrict__ src,
                                          int nhalf, int tid, int nthr) {
  int nchunk = nhalf >> 3;  // 8 halves = 16B
  for (int c = tid; c < nchunk; c += nthr)
    ((uint4*)dst)[c] = ((const uint4*)src)[c];
}

// ---------------- weight packing ----------------
// Packed layout per layer: [kslab][ntile][lane 0..31][elem 0..15] matching the
// 16-bit A/B WMMA fragment striping (lane<16: K {0..7,16..23}; lane>=16: {8..15,24..31}).
__global__ void pack_w_kernel(const float* __restrict__ W, _Float16* __restrict__ out,
                              int K, int total) {
  int i = blockIdx.x * 256 + threadIdx.x;
  if (i >= total) return;
  int per = K * DF;
  int layer = i / per, r = i % per;
  int elem = r & 15, t2 = r >> 4;
  int lane = t2 & 31, frag = t2 >> 5;
  int ntile = frag & 3, kslab = frag >> 2;
  int l = lane & 15, hi = lane >> 4;
  int k = kslab * 32 + (elem < 8 ? hi * 8 + elem : 16 + hi * 8 + (elem - 8));
  int n = ntile * 16 + l;
  out[i] = (_Float16)W[(size_t)layer * per + (size_t)k * DF + n];
}

// ---------------- small elementwise kernels ----------------

__global__ void zero_f32_kernel(float* p, int n) {
  int i = blockIdx.x * 256 + threadIdx.x;
  if (i < n) p[i] = 0.f;
}

__global__ void copy2_kernel(const float* __restrict__ x, const float* __restrict__ s,
                             float* __restrict__ xb, float* __restrict__ sb, int n) {
  int i = blockIdx.x * 256 + threadIdx.x;
  if (i < n) { xb[i] = x[i]; sb[i] = s[i]; }
}

__global__ void degree_kernel(const int* __restrict__ ei, int E, float* __restrict__ deg) {
  int e = blockIdx.x * 256 + threadIdx.x;
  if (e < E) atomicAdd(&deg[ei[E + e]], 1.0f);
}

__global__ void dinv_kernel(const float* __restrict__ deg, float* __restrict__ dinv, int n) {
  int i = blockIdx.x * 256 + threadIdx.x;
  if (i < n) dinv[i] = rsqrtf(deg[i] + 1.0f);   // indegree + self loop, always > 0
}

__global__ void concat_kernel(const float* __restrict__ a, const float* __restrict__ b,
                              float* __restrict__ out, int n128) {
  int i = blockIdx.x * 256 + threadIdx.x;
  if (i >= n128) return;
  int v = i >> 7, j = i & 127;
  out[i] = (j < DF) ? a[(size_t)v * DF + j] : b[(size_t)v * DF + (j - DF)];
}

// ---------------- edge scatter kernels ----------------

// agg[dst, 0:128] += [x[src] | s[src]]  (one wave per edge, float4 lanes)
__global__ void gin_scatter_kernel(const int* __restrict__ ei, int E,
                                   const float* __restrict__ xb, const float* __restrict__ sb,
                                   float* __restrict__ agg) {
  int t = blockIdx.x * 256 + threadIdx.x;
  int e = t >> 5, lane = t & 31;
  if (e >= E) return;
  int src = ei[e], dst = ei[E + e];
  const float* base = (lane < 16) ? (xb + (size_t)src * DF + lane * 4)
                                  : (sb + (size_t)src * DF + (lane - 16) * 4);
  float4 v = *(const float4*)base;
  float* ad = agg + (size_t)dst * 128 + lane * 4;
  atomicAdd(ad + 0, v.x); atomicAdd(ad + 1, v.y);
  atomicAdd(ad + 2, v.z); atomicAdd(ad + 3, v.w);
}

// sacc[dst] += dinv[src]*dinv[dst] * sw[src]   (16 lanes/edge, float4)
__global__ void gcn_scatter_kernel(const int* __restrict__ ei, int E,
                                   const float* __restrict__ sw, const float* __restrict__ dinv,
                                   float* __restrict__ sacc) {
  int t = blockIdx.x * 256 + threadIdx.x;
  int e = t >> 4, lane = t & 15;
  if (e >= E) return;
  int src = ei[e], dst = ei[E + e];
  float nrm = dinv[src] * dinv[dst];
  float4 v = *(const float4*)(sw + (size_t)src * DF + lane * 4);
  float* ad = sacc + (size_t)dst * DF + lane * 4;
  atomicAdd(ad + 0, nrm * v.x); atomicAdd(ad + 1, nrm * v.y);
  atomicAdd(ad + 2, nrm * v.z); atomicAdd(ad + 3, nrm * v.w);
}

__global__ void gcn_init_kernel(const float* __restrict__ sw, const float* __restrict__ dinv,
                                const float* __restrict__ bg, float* __restrict__ sacc, int n64) {
  int i = blockIdx.x * 256 + threadIdx.x;
  if (i >= n64) return;
  int v = i >> 6, f = i & 63;
  float d = dinv[v];
  sacc[i] = d * d * sw[i] + bg[f];
}

__global__ void tanh_kernel(const float* __restrict__ a, float* __restrict__ o, int n) {
  int i = blockIdx.x * 256 + threadIdx.x;
  if (i < n) o[i] = tanhf(a[i]);
}

// ---------------- WMMA GEMM kernels ----------------
// 256 threads = 8 waves; each wave owns a 16-row stripe and all 4 column tiles,
// so one A fragment feeds 4 back-to-back v_wmma. Weights staged in LDS,
// fragment-major, read as ds_load_b128 pairs.

// Fused GIN MLP: x = leaky(agg[N,128] @ W1[128,64]) @ W2[64,64]
__global__ __launch_bounds__(256) void gin_mlp_kernel(const float* __restrict__ agg,
                                                      const _Float16* __restrict__ W1p,
                                                      const _Float16* __restrict__ W2p,
                                                      float* __restrict__ xout, int M) {
  __shared__ __align__(32) _Float16 sW1[4 * 4 * 32 * 16];   // 16KB: K=128
  __shared__ __align__(32) _Float16 sW2[2 * 4 * 32 * 16];   //  8KB: K=64
  __shared__ __align__(32) _Float16 T[8][16][72];           // 18KB: per-wave f16 intermediate
  int tid = threadIdx.x;
  stage_lds(sW1, W1p, 4 * 4 * 32 * 16, tid, 256);
  stage_lds(sW2, W2p, 2 * 4 * 32 * 16, tid, 256);
  __syncthreads();

  int lane = tid & 31, wave = tid >> 5;
  int l = lane & 15, hi = lane >> 4;
  int m0 = (blockIdx.x * 8 + wave) * 16;
  if (m0 >= M) return;   // wave-uniform

  v8f acc[4] = {};
#pragma unroll
  for (int kslab = 0; kslab < 4; ++kslab) {
    v16h a = load_a_f32(agg, 128, m0, kslab * 32, lane);
#pragma unroll
    for (int nt = 0; nt < 4; ++nt)
      acc[nt] = wmma16(a, load_b_lds(sW1, kslab, nt, lane), acc[nt]);
  }
  _Float16* Tw = &T[wave][0][0];
#pragma unroll
  for (int nt = 0; nt < 4; ++nt)
#pragma unroll
    for (int r = 0; r < 8; ++r) {
      float v = acc[nt][r];
      v = v > 0.f ? v : 0.01f * v;            // leaky relu
      Tw[(size_t)(r + hi * 8) * 72 + nt * 16 + l] = (_Float16)v;
    }
  // per-wave private tile: no cross-wave barrier needed (dscnt handled by compiler)
  v8f acc2[4] = {};
#pragma unroll
  for (int kslab = 0; kslab < 2; ++kslab) {
    int k0 = kslab * 32;
    v16h a;
#pragma unroll
    for (int j = 0; j < 8; ++j) a[j] = Tw[(size_t)l * 72 + k0 + hi * 8 + j];
#pragma unroll
    for (int j = 0; j < 8; ++j) a[8 + j] = Tw[(size_t)l * 72 + k0 + 16 + hi * 8 + j];
#pragma unroll
    for (int nt = 0; nt < 4; ++nt)
      acc2[nt] = wmma16(a, load_b_lds(sW2, kslab, nt, lane), acc2[nt]);
  }
#pragma unroll
  for (int nt = 0; nt < 4; ++nt)
#pragma unroll
    for (int r = 0; r < 8; ++r)
      xout[(size_t)(m0 + r + hi * 8) * DF + nt * 16 + l] = acc2[nt][r];
}

// C[N,64] = A[N,64] @ B[64,64]
__global__ __launch_bounds__(256) void gemm64_kernel(const float* __restrict__ A,
                                                     const _Float16* __restrict__ Bp,
                                                     float* __restrict__ C, int M) {
  __shared__ __align__(32) _Float16 sW[2 * 4 * 32 * 16];    // 8KB
  int tid = threadIdx.x;
  stage_lds(sW, Bp, 2 * 4 * 32 * 16, tid, 256);
  __syncthreads();

  int lane = tid & 31, wave = tid >> 5;
  int l = lane & 15, hi = lane >> 4;
  int m0 = (blockIdx.x * 8 + wave) * 16;
  if (m0 >= M) return;

  v8f acc[4] = {};
#pragma unroll
  for (int kslab = 0; kslab < 2; ++kslab) {
    v16h a = load_a_f32(A, DF, m0, kslab * 32, lane);
#pragma unroll
    for (int nt = 0; nt < 4; ++nt)
      acc[nt] = wmma16(a, load_b_lds(sW, kslab, nt, lane), acc[nt]);
  }
#pragma unroll
  for (int nt = 0; nt < 4; ++nt)
#pragma unroll
    for (int r = 0; r < 8; ++r)
      C[(size_t)(m0 + r + hi * 8) * DF + nt * 16 + l] = acc[nt][r];
}

// pooled[batch[row]] += hfin[N,128] @ Wh[128,64] + bh   (fused global_add_pool)
__global__ __launch_bounds__(256) void final_gemm_pool_kernel(const float* __restrict__ hfin,
                                                              const _Float16* __restrict__ Whp,
                                                              const float* __restrict__ bh,
                                                              const int* __restrict__ batch,
                                                              float* __restrict__ pooled, int M) {
  __shared__ __align__(32) _Float16 sW[4 * 4 * 32 * 16];    // 16KB
  int tid = threadIdx.x;
  stage_lds(sW, Whp, 4 * 4 * 32 * 16, tid, 256);
  __syncthreads();

  int lane = tid & 31, wave = tid >> 5;
  int l = lane & 15, hi = lane >> 4;
  int m0 = (blockIdx.x * 8 + wave) * 16;
  if (m0 >= M) return;

  v8f acc[4] = {};
#pragma unroll
  for (int kslab = 0; kslab < 4; ++kslab) {
    v16h a = load_a_f32(hfin, 128, m0, kslab * 32, lane);
#pragma unroll
    for (int nt = 0; nt < 4; ++nt)
      acc[nt] = wmma16(a, load_b_lds(sW, kslab, nt, lane), acc[nt]);
  }
#pragma unroll
  for (int nt = 0; nt < 4; ++nt) {
    int col = nt * 16 + l;
    float bias = bh[col];
#pragma unroll
    for (int r = 0; r < 8; ++r) {
      int row = m0 + r + hi * 8;
      int g = batch[row];
      atomicAdd(&pooled[(size_t)g * DF + col], acc[nt][r] + bias);
    }
  }
}

// ---------------- batch norm ----------------

__global__ __launch_bounds__(256) void bn_reduce_kernel(const float* __restrict__ x, int N,
                                                        float* __restrict__ bnsum,
                                                        float* __restrict__ bnsq) {
  __shared__ float s1[256], s2[256];
  int f = threadIdx.x & 63, g = threadIdx.x >> 6;
  float a = 0.f, b = 0.f;
  for (int r = blockIdx.x * 4 + g; r < N; r += gridDim.x * 4) {
    float v = x[(size_t)r * DF + f];
    a += v; b += v * v;
  }
  s1[threadIdx.x] = a; s2[threadIdx.x] = b;
  __syncthreads();
  if (threadIdx.x < 64) {
    float ta = s1[threadIdx.x] + s1[threadIdx.x + 64] + s1[threadIdx.x + 128] + s1[threadIdx.x + 192];
    float tb = s2[threadIdx.x] + s2[threadIdx.x + 64] + s2[threadIdx.x + 128] + s2[threadIdx.x + 192];
    atomicAdd(&bnsum[threadIdx.x], ta);
    atomicAdd(&bnsq[threadIdx.x], tb);
  }
}

__global__ void bn_apply_kernel(const float* __restrict__ x, const float* __restrict__ bnsum,
                                const float* __restrict__ bnsq, const float* __restrict__ gamma,
                                const float* __restrict__ beta, float* __restrict__ xl,
                                int N, int n64) {
  int i = blockIdx.x * 256 + threadIdx.x;
  if (i >= n64) return;
  int f = i & 63;
  float invN = 1.0f / (float)N;
  float mean = bnsum[f] * invN;
  float var = bnsq[f] * invN - mean * mean;
  xl[i] = gamma[f] * (x[i] - mean) * rsqrtf(var + BN_EPS) + beta[f];
}

// ---------------- host launch ----------------

static inline int cdiv(int a, int b) { return (a + b - 1) / b; }

extern "C" void kernel_launch(void* const* d_in, const int* in_sizes, int n_in,
                              void* d_out, int out_size, void* d_ws, size_t ws_size,
                              hipStream_t stream) {
  const float* x_in  = (const float*)d_in[0];
  const float* s_in  = (const float*)d_in[1];
  const float* W1    = (const float*)d_in[2];
  const float* W2    = (const float*)d_in[3];
  const float* gamma = (const float*)d_in[4];
  const float* beta  = (const float*)d_in[5];
  const float* Wg    = (const float*)d_in[6];
  const float* bg    = (const float*)d_in[7];
  const float* Wh    = (const float*)d_in[8];
  const float* bh    = (const float*)d_in[9];
  const int*   ei    = (const int*)d_in[10];
  const int*   batch = (const int*)d_in[11];

  const int N = in_sizes[0] / DF;
  const int E = in_sizes[10] / 2;
  const int Lnum = in_sizes[2] / (2 * DF * DF);
  const int G = out_size / DF - N;

  float* pooled = (float*)d_out;                   // [G,64]
  float* xlocal = (float*)d_out + (size_t)G * DF;  // [N,64]

  // workspace layout
  float* agg  = (float*)d_ws;            // [N,128]  (reused as sacc / hfin)
  float* xbuf = agg  + (size_t)N * 128;  // [N,64]
  float* sbuf = xbuf + (size_t)N * DF;   // [N,64]
  float* swb  = sbuf + (size_t)N * DF;   // [N,64]
  float* deg  = swb  + (size_t)N * DF;   // [N]
  float* dinv = deg  + N;                // [N]
  float* bnsum = dinv + N;               // [64]
  float* bnsq  = bnsum + 64;             // [64]
  uintptr_t pk = ((uintptr_t)(bnsq + 64) + 31) & ~(uintptr_t)31;  // 32B align packed weights
  _Float16* W1p = (_Float16*)pk;                  // [L][4][4][32][16]
  _Float16* W2p = W1p + (size_t)Lnum * 128 * DF;  // [L][2][4][32][16]
  _Float16* Wgp = W2p + (size_t)Lnum * DF * DF;
  _Float16* Whp = Wgp + (size_t)Lnum * DF * DF;   // [4][4][32][16]
  float* sacc = agg;

  const int n64 = N * DF, n128 = N * 128;
  const int mb = cdiv(cdiv(N, 16), 8);   // GEMM blocks: 8 waves x 16 rows

  // re-initialize all accumulators every call (graph-replay safe)
  zero_f32_kernel<<<cdiv(G * DF, 256), 256, 0, stream>>>(pooled, G * DF);
  zero_f32_kernel<<<cdiv(N, 256), 256, 0, stream>>>(deg, N);

  // pack weights into WMMA fragment-major f16 layout
  pack_w_kernel<<<cdiv(Lnum * 128 * DF, 256), 256, 0, stream>>>(W1, W1p, 128, Lnum * 128 * DF);
  pack_w_kernel<<<cdiv(Lnum * DF * DF, 256), 256, 0, stream>>>(W2, W2p, DF, Lnum * DF * DF);
  pack_w_kernel<<<cdiv(Lnum * DF * DF, 256), 256, 0, stream>>>(Wg, Wgp, DF, Lnum * DF * DF);
  pack_w_kernel<<<cdiv(128 * DF, 256), 256, 0, stream>>>(Wh, Whp, 128, 128 * DF);

  copy2_kernel<<<cdiv(n64, 256), 256, 0, stream>>>(x_in, s_in, xbuf, sbuf, n64);

  // GCN degree normalization (with self loops)
  degree_kernel<<<cdiv(E, 256), 256, 0, stream>>>(ei, E, deg);
  dinv_kernel<<<cdiv(N, 256), 256, 0, stream>>>(deg, dinv, N);

  for (int i = 0; i < Lnum; ++i) {
    // GIN: agg = [x|s] + scatter_sum([x|s][src] -> dst)
    concat_kernel<<<cdiv(n128, 256), 256, 0, stream>>>(xbuf, sbuf, agg, n128);
    gin_scatter_kernel<<<cdiv(E * 32, 256), 256, 0, stream>>>(ei, E, xbuf, sbuf, agg);
    // fused MLP (WMMA): xbuf = leaky(agg@W1)@W2
    gin_mlp_kernel<<<mb, 256, 0, stream>>>(agg, W1p + (size_t)i * 128 * DF,
                                           W2p + (size_t)i * DF * DF, xbuf, N);
    if (i == Lnum - 1) {  // only last layer's x_local is observable
      zero_f32_kernel<<<1, 256, 0, stream>>>(bnsum, 128);
      bn_reduce_kernel<<<512, 256, 0, stream>>>(xbuf, N, bnsum, bnsq);
      bn_apply_kernel<<<cdiv(n64, 256), 256, 0, stream>>>(xbuf, bnsum, bnsq,
          gamma + (size_t)i * DF, beta + (size_t)i * DF, xlocal, N, n64);
    }
    // GCN: s = tanh(D^-1/2 (A+I) D^-1/2 (s@Wg) + bg)
    gemm64_kernel<<<mb, 256, 0, stream>>>(sbuf, Wgp + (size_t)i * DF * DF, swb, N);
    gcn_init_kernel<<<cdiv(n64, 256), 256, 0, stream>>>(swb, dinv, bg + (size_t)i * DF, sacc, n64);
    gcn_scatter_kernel<<<cdiv(E * 16, 256), 256, 0, stream>>>(ei, E, swb, dinv, sacc);
    tanh_kernel<<<cdiv(n64, 256), 256, 0, stream>>>(sacc, sbuf, n64);
  }

  // out = [x_local | s] @ Wh + bh, pooled by graph id (fused)
  concat_kernel<<<cdiv(n128, 256), 256, 0, stream>>>(xlocal, sbuf, agg, n128);
  final_gemm_pool_kernel<<<mb, 256, 0, stream>>>(agg, Whp, bh, batch, pooled, N);
}
